// _global_attention_sop_fs_22814866277109
// MI455X (gfx1250) — matrix operand order repaired and checked by reading, not compiled, not run on hardware
//
#include <hip/hip_runtime.h>
#include <hip/hip_bf16.h>
#include <math.h>

#define D_DIM   256
#define M_NODES 200000
#define B_GRAPHS 128
#define NTILES  (M_NODES / 16)          // 12500 exact 16-row tiles

typedef __attribute__((ext_vector_type(16))) _Float16 v16h;
typedef __attribute__((ext_vector_type(8)))  float    v8f;

// ---------------------------------------------------------------------------
// float atomic max via signed/unsigned integer monotone mapping
// ---------------------------------------------------------------------------
__device__ __forceinline__ void atomicMaxF32(float* addr, float val) {
    if (val >= 0.0f) {
        atomicMax((int*)addr, __float_as_int(val));
    } else {
        atomicMin((unsigned int*)addr, __float_as_uint(val));
    }
}

// ---------------------------------------------------------------------------
// 1) init: zero output, init segment max/sum
// ---------------------------------------------------------------------------
__global__ void init_kernel(float* __restrict__ out,
                            float* __restrict__ maxbuf,
                            float* __restrict__ sumbuf) {
    int i = blockIdx.x * blockDim.x + threadIdx.x;
    if (i < B_GRAPHS * D_DIM) out[i] = 0.0f;
    if (i < B_GRAPHS) {
        maxbuf[i] = -__builtin_huge_valf();
        sumbuf[i] = 0.0f;
    }
}

// ---------------------------------------------------------------------------
// 2) W[k*256+n] (f32, row-major D x D) -> Wt[n*256+k] (f16, transposed)
//    so that a B-fragment column n has contiguous k in LDS.
// ---------------------------------------------------------------------------
__global__ void convert_w_kernel(const float* __restrict__ W,
                                 _Float16* __restrict__ Wt) {
    int i = blockIdx.x * blockDim.x + threadIdx.x;   // i = k*256 + n
    int k = i >> 8;
    int n = i & 255;
    Wt[n * D_DIM + k] = (_Float16)W[i];
}

// ---------------------------------------------------------------------------
// 3) imp[m] = x_m @ W @ x_m^T  via v_wmma_f32_16x16x32_f16.
//    One wave per 16-row tile; full Wt (128 KB f16) resident in LDS.
// ---------------------------------------------------------------------------
__global__ void __launch_bounds__(256)
gemm_imp_kernel(const float* __restrict__ x,
                const _Float16* __restrict__ Wt,
                float* __restrict__ imp) {
    extern __shared__ char smem_raw[];
    unsigned int* sWu = (unsigned int*)smem_raw;     // Wt as packed f16 pairs

    // cooperative fill of LDS with Wt (32768 dwords)
    const unsigned int* wsrc = (const unsigned int*)Wt;
    for (int i = threadIdx.x; i < (D_DIM * D_DIM / 2); i += 256) sWu[i] = wsrc[i];
    __syncthreads();

    const int lane     = threadIdx.x & 31;
    const int col      = lane & 15;      // M (for A) / N (for B) / N (for C)
    const int laneHalf = lane >> 4;      // selects K half (contiguous k order used here)
    const int globalWave = blockIdx.x * 8 + (threadIdx.x >> 5);
    const int totalWaves = gridDim.x * 8;

    for (int t = globalWave; t < NTILES; t += totalWaves) {
        const int rowbase = t * 16;
        const float* xrow = x + (size_t)(rowbase + col) * D_DIM;

        // --- A fragments for all 8 k-blocks (16x32 f16 each) -------------
        // layout: lanes 0-15 hold K = 0..15 of the k-block,
        //         lanes 16-31 hold K = 16..31; 2 f16 per VGPR.
        union { v16h h; _Float16 e[16]; } A[8];
        #pragma unroll
        for (int kb = 0; kb < 8; ++kb) {
            const int kbase = kb * 32 + laneHalf * 16;
            #pragma unroll
            for (int v = 0; v < 8; ++v) {
                const int k = kbase + 2 * v;
                float2 xv = *(const float2*)(xrow + k);
                A[kb].e[2 * v]     = (_Float16)xv.x;
                A[kb].e[2 * v + 1] = (_Float16)xv.y;
            }
        }

        float partial[8] = {0.f, 0.f, 0.f, 0.f, 0.f, 0.f, 0.f, 0.f};

        // --- 16 column tiles of Y = X @ W --------------------------------
        #pragma unroll 1
        for (int jb = 0; jb < 16; ++jb) {
            const int n = jb * 16 + col;
            const unsigned int* wcol = sWu + ((size_t)n * D_DIM) / 2;

            v8f acc = {0.f, 0.f, 0.f, 0.f, 0.f, 0.f, 0.f, 0.f};
            #pragma unroll
            for (int kb = 0; kb < 8; ++kb) {
                // B layout (32x16): lanes 0-15 hold K=0..15 of the k-block,
                // lanes 16-31 hold K=16..31; VGPR v = packed {2v, 2v+1}.
                union { v16h h; unsigned int u[8]; } Bf;
                const int kdw = (kb * 32 + laneHalf * 16) >> 1;  // dword index
                #pragma unroll
                for (int v = 0; v < 8; ++v) Bf.u[v] = wcol[kdw + v];

                acc = __builtin_amdgcn_wmma_f32_16x16x32_f16(
                        false, A[kb].h, false, Bf.h,
                        (short)0, acc, false, false);
            }

            // rowsum(Y .* X): C layout -> row = rowbase + r + 8*laneHalf,
            // column = jb*16 + col
            const float* xcol =
                x + (size_t)(rowbase + 8 * laneHalf) * D_DIM + jb * 16 + col;
            #pragma unroll
            for (int r = 0; r < 8; ++r)
                partial[r] += acc[r] * xcol[(size_t)r * D_DIM];
        }

        // reduce across the 16 lanes of each half, write imp
        #pragma unroll
        for (int r = 0; r < 8; ++r) {
            float s = partial[r];
            s += __shfl_xor(s, 1, 32);
            s += __shfl_xor(s, 2, 32);
            s += __shfl_xor(s, 4, 32);
            s += __shfl_xor(s, 8, 32);
            if (col == 0) imp[rowbase + 8 * laneHalf + r] = s;
        }
    }
}

// ---------------------------------------------------------------------------
// 4) per-segment max of imp (softmax is shift invariant -> bias b dropped)
// ---------------------------------------------------------------------------
__global__ void segmax_kernel(const float* __restrict__ imp,
                              const int* __restrict__ batch,
                              float* __restrict__ maxbuf) {
    int i = blockIdx.x * blockDim.x + threadIdx.x;
    if (i < M_NODES) atomicMaxF32(&maxbuf[batch[i]], imp[i]);
}

// ---------------------------------------------------------------------------
// 5) z = exp(imp - max[g]) in place; accumulate segment sums
// ---------------------------------------------------------------------------
__global__ void expsum_kernel(float* __restrict__ imp_z,
                              const int* __restrict__ batch,
                              const float* __restrict__ maxbuf,
                              float* __restrict__ sumbuf) {
    int i = blockIdx.x * blockDim.x + threadIdx.x;
    if (i < M_NODES) {
        int g = batch[i];
        float z = __expf(imp_z[i] - maxbuf[g]);
        imp_z[i] = z;
        atomicAdd(&sumbuf[g], z);
    }
}

// ---------------------------------------------------------------------------
// 6) out[g,:] += (z[m]/sum[g]) * x[m,:].  batch is sorted: one block owns a
//    contiguous 256-row chunk, thread = column, flush on segment change.
// ---------------------------------------------------------------------------
__global__ void __launch_bounds__(256)
pool_kernel(const float* __restrict__ x,
            const int* __restrict__ batch,
            const float* __restrict__ z,
            const float* __restrict__ sumbuf,
            float* __restrict__ out) {
    __shared__ int   sg[256];
    __shared__ float sw[256];

    const int m0 = blockIdx.x * 256;
    const int d  = threadIdx.x;

    const int mi = m0 + d;
    if (mi < M_NODES) {
        int g = batch[mi];
        sg[d] = g;
        sw[d] = z[mi] / sumbuf[g];
    }
    __syncthreads();

    const int nrows = min(256, M_NODES - m0);
    float acc = 0.0f;
    int gcur = sg[0];
    for (int j = 0; j < nrows; ++j) {
        int g = sg[j];
        if (g != gcur) {                       // uniform across the block
            atomicAdd(&out[gcur * D_DIM + d], acc);
            acc = 0.0f;
            gcur = g;
        }
        acc += sw[j] * x[(size_t)(m0 + j) * D_DIM + d];
    }
    atomicAdd(&out[gcur * D_DIM + d], acc);
}

// ---------------------------------------------------------------------------
extern "C" void kernel_launch(void* const* d_in, const int* in_sizes, int n_in,
                              void* d_out, int out_size, void* d_ws, size_t ws_size,
                              hipStream_t stream) {
    (void)in_sizes; (void)n_in; (void)out_size; (void)ws_size;

    const float* x     = (const float*)d_in[0];
    const int*   batch = (const int*)d_in[1];
    // d_in[2] = edge  : dead in the reference (degree branch unused) -> skipped
    const float* W     = (const float*)d_in[3];
    // d_in[4] = b     : softmax is shift invariant -> no effect on out
    float* out = (float*)d_out;

    char* ws = (char*)d_ws;
    _Float16* Wt    = (_Float16*)(ws);             // 131072 B
    float*    imp_z = (float*)(ws + 131072);       // 800000 B (imp, then z)
    float*    maxb  = (float*)(ws + 931072);       // 512 B
    float*    sumb  = (float*)(ws + 931584);       // 512 B

    init_kernel<<<(B_GRAPHS * D_DIM + 255) / 256, 256, 0, stream>>>(out, maxb, sumb);
    convert_w_kernel<<<(D_DIM * D_DIM) / 256, 256, 0, stream>>>(W, Wt);

    // 640 WGs x 8 waves = 5120 waves over 12500 tiles; 128 KB LDS for Wt
    gemm_imp_kernel<<<640, 256, 131072, stream>>>(x, Wt, imp_z);

    const int mb = (M_NODES + 255) / 256;
    segmax_kernel<<<mb, 256, 0, stream>>>(imp_z, batch, maxb);
    expsum_kernel<<<mb, 256, 0, stream>>>(imp_z, batch, maxb, sumb);
    pool_kernel  <<<mb, 256, 0, stream>>>(x, batch, imp_z, sumb, out);
}
